// TENLayer_7017976561859
// MI455X (gfx1250) — matrix operand order repaired and checked by reading, not compiled, not run on hardware
//
#include <hip/hip_runtime.h>
#include <cstdint>
#include <cstddef>

typedef __bf16 bf16_t;
typedef __attribute__((ext_vector_type(8)))  __bf16 v8bf;
typedef __attribute__((ext_vector_type(16))) __bf16 v16bf;
typedef __attribute__((ext_vector_type(8)))  float  v8f;
typedef __attribute__((ext_vector_type(4)))  unsigned int v4u;
typedef __attribute__((ext_vector_type(8)))  int v8i;
typedef __attribute__((ext_vector_type(4)))  int v4i;

#define BT_TOK 16384   // B*T tokens
#define DMODEL 1024
#define KD     64
#define MLPD   4096
#define TLEN   4096
#define BATCH  4
#define NCH    32      // scan chunks per sequence
#define CHLEN  128     // TLEN/NCH

#define LDS_PITCH 40   // 32 k-elements + 8 pad (80B rows -> conflict-free 16B frag loads)

#if defined(__HIP_DEVICE_COMPILE__) && __has_builtin(__builtin_amdgcn_tensor_load_to_lds) && __has_builtin(__builtin_amdgcn_s_wait_tensorcnt)
#define HAVE_TDM 1
#else
#define HAVE_TDM 0
#endif

// ---------------- WMMA fragment loads (per CDNA5 ISA 16-bit layouts) ----------------
// A 16x32: lane L holds row L%16; K = {g*8..g*8+7, 16+g*8..+7}, g = L/16
__device__ __forceinline__ v16bf load_frag_a(const bf16_t* p) {
  v8bf lo = *reinterpret_cast<const v8bf*>(p);
  v8bf hi = *reinterpret_cast<const v8bf*>(p + 16);
  v16bf r;
#pragma unroll
  for (int i = 0; i < 8; ++i) { r[i] = lo[i]; r[i + 8] = hi[i]; }
  return r;
}
// B 32x16 (stored N-major / pre-transposed): lane L holds col L%16; K = {g*8..g*8+7, 16+g*8..+7}
__device__ __forceinline__ v16bf load_frag_b(const bf16_t* p) {
  v8bf lo = *reinterpret_cast<const v8bf*>(p);
  v8bf hi = *reinterpret_cast<const v8bf*>(p + 16);
  v16bf r;
#pragma unroll
  for (int i = 0; i < 8; ++i) { r[i] = lo[i]; r[i + 8] = hi[i]; }
  return r;
}

enum { EPI_F32 = 0, EPI_BF16 = 1, EPI_BIAS_RESID_F32 = 2, EPI_BIAS_GELU_BF16 = 3 };

template <int EPI>
__device__ __forceinline__ void gemm_epilogue(v8f (&accs)[2][2], int m0, int n0, int g, int rsel,
                                              float* __restrict__ Cf, bf16_t* __restrict__ Cb, int ldc,
                                              const float* __restrict__ bias,
                                              const float* __restrict__ resid, int ldr) {
#pragma unroll
  for (int i = 0; i < 2; ++i) {
#pragma unroll
    for (int j = 0; j < 2; ++j) {
      const int rowb = m0 + i * 16 + g * 8;   // C/D layout: VGPR r -> M = r + 8*(lane/16)
      const int col  = n0 + j * 16 + rsel;    // lane%16 -> N
#pragma unroll
      for (int r = 0; r < 8; ++r) {
        float v = accs[i][j][r];
        const size_t off = (size_t)(rowb + r) * ldc + col;
        if constexpr (EPI == EPI_F32) {
          Cf[off] = v;
        } else if constexpr (EPI == EPI_BF16) {
          Cb[off] = (bf16_t)v;
        } else if constexpr (EPI == EPI_BIAS_RESID_F32) {
          Cf[off] = v + bias[col] + resid[(size_t)(rowb + r) * ldr + col];
        } else { // bias + exact gelu -> bf16
          float t = v + bias[col];
          Cb[off] = (bf16_t)(0.5f * t * (1.f + erff(t * 0.70710678118654752f)));
        }
      }
    }
  }
}

// -------- direct (global->VGPR) GEMM: used for small/folded GEMMs --------
// C(MxN) = A(MxK bf16 row-major) x B(KxN) with Bt = B^T (NxK row-major).
// Block tile 128x64, 8 waves (4x2), wave = 2x2 WMMA tiles, K-step 32.
template <int EPI>
__global__ __launch_bounds__(256, 2)
void gemm_bf16_wmma(const bf16_t* __restrict__ A, int lda,
                    const bf16_t* __restrict__ Bt, int ldb,
                    float* __restrict__ Cf, bf16_t* __restrict__ Cb, int ldc,
                    const float* __restrict__ bias,
                    const float* __restrict__ resid, int ldr,
                    int Kdim) {
  const int lane = threadIdx.x & 31;
  const int wave = threadIdx.x >> 5;
  const int wm = wave & 3, wn = wave >> 2;
  const int m0 = blockIdx.y * 128 + wm * 32;
  const int n0 = blockIdx.x * 64 + wn * 32;
  const int rsel = lane & 15;
  const int g = lane >> 4;

  v8f acc00 = {}, acc01 = {}, acc10 = {}, acc11 = {};
  const bf16_t* a0 = A + (size_t)(m0 + rsel) * lda + g * 8;
  const bf16_t* a1 = a0 + (size_t)16 * lda;
  const bf16_t* b0 = Bt + (size_t)(n0 + rsel) * ldb + g * 8;
  const bf16_t* b1 = b0 + (size_t)16 * ldb;

  for (int k0 = 0; k0 < Kdim; k0 += 32) {
    v16bf fa0 = load_frag_a(a0 + k0);
    v16bf fa1 = load_frag_a(a1 + k0);
    v16bf fb0 = load_frag_b(b0 + k0);
    v16bf fb1 = load_frag_b(b1 + k0);
    acc00 = __builtin_amdgcn_wmma_f32_16x16x32_bf16(false, fa0, false, fb0, (short)0, acc00, false, false);
    acc01 = __builtin_amdgcn_wmma_f32_16x16x32_bf16(false, fa0, false, fb1, (short)0, acc01, false, false);
    acc10 = __builtin_amdgcn_wmma_f32_16x16x32_bf16(false, fa1, false, fb0, (short)0, acc10, false, false);
    acc11 = __builtin_amdgcn_wmma_f32_16x16x32_bf16(false, fa1, false, fb1, (short)0, acc11, false, false);
  }
  v8f accs[2][2] = {{acc00, acc01}, {acc10, acc11}};
  gemm_epilogue<EPI>(accs, m0, n0, g, rsel, Cf, Cb, ldc, bias, resid, ldr);
}

// -------- Tensor Data Mover tile fetch: 2-D tile (32 k-elements x rows) -> LDS --------
// D# per CDNA5 ISA ch.8: group0 {count=1, lds_addr, 57b global addr, type=2},
// group1 {data_size=2B, pad_enable, pad_interval=16dw, pad_amount=4dw (-> 80B LDS pitch),
//         tensor_dim0/1, tile_dim0=32, tile_dim1=rows, dim0 stride}.
// This toolchain's builtin takes 6 args: (g0 u32x4, g1 i32x8, g2 i32x4, g3 i32x4, i32x8, cpol).
__device__ __forceinline__ void tdm_load_tile(unsigned lds_addr, const bf16_t* gsrc,
                                              int tile_rows, int tensor_k, int tensor_rows,
                                              long stride_elems) {
#if HAVE_TDM
  const unsigned long long ga = (unsigned long long)(uintptr_t)gsrc;
  const unsigned long long str = (unsigned long long)stride_elems;
  v4u g0;
  g0[0] = 1u;                                           // count=1 (valid descriptor)
  g0[1] = lds_addr;                                     // lds_addr [63:32]
  g0[2] = (unsigned)(ga & 0xFFFFFFFFu);                 // global_addr lo
  g0[3] = (unsigned)((ga >> 32) & 0x01FFFFFFu) | 0x80000000u;  // global_addr hi + type=2
  v8i g1;
  g1[0] = (int)((1u << 16) | (1u << 20) | (3u << 22) | (3u << 25)); // data_size=2B, pad en, 16dw interval, 4dw pad
  g1[1] = (int)(((unsigned)tensor_k & 0xFFFFu) << 16);              // tensor_dim0 lo16
  g1[2] = (int)((((unsigned)tensor_k >> 16) & 0xFFFFu) |
                (((unsigned)tensor_rows & 0xFFFFu) << 16));         // tensor_dim0 hi | tensor_dim1 lo
  g1[3] = (int)(((((unsigned)tensor_rows) >> 16) & 0xFFFFu) |
                (32u << 16));                                       // tensor_dim1 hi | tile_dim0=32
  g1[4] = (int)((unsigned)tile_rows & 0xFFFFu);                     // tile_dim1 | tile_dim2=0
  g1[5] = (int)(unsigned)(str & 0xFFFFFFFFu);                       // dim0_stride lo32
  g1[6] = (int)(unsigned)((str >> 32) & 0xFFFFu);                   // dim0_stride hi16 | dim1_stride lo=0
  g1[7] = 0;
  const v4i z4 = {0, 0, 0, 0};
  const v8i z8 = {0, 0, 0, 0, 0, 0, 0, 0};
  __builtin_amdgcn_tensor_load_to_lds(g0, g1, z4, z4, z8, 0);
#else
  (void)lds_addr; (void)gsrc; (void)tile_rows; (void)tensor_k; (void)tensor_rows; (void)stride_elems;
#endif
}

// sync fallback staging (also used by host pass)
__device__ __forceinline__ void stage_tiles_sync(bf16_t* Abuf, bf16_t* Bbuf,
                                                 const bf16_t* Ag, const bf16_t* Bg,
                                                 int lda, int ldb, int tid) {
#pragma unroll
  for (int i = 0; i < 3; ++i) {          // 512 A-chunks + 256 B-chunks of 16B, 256 threads
    const int idx = tid + i * 256;
    if (idx < 512) {
      const int row = idx >> 2, ch = idx & 3;
      *reinterpret_cast<v8bf*>(Abuf + row * LDS_PITCH + ch * 8) =
          *reinterpret_cast<const v8bf*>(Ag + (size_t)row * lda + ch * 8);
    } else {
      const int j = idx - 512, row = j >> 2, ch = j & 3;
      *reinterpret_cast<v8bf*>(Bbuf + row * LDS_PITCH + ch * 8) =
          *reinterpret_cast<const v8bf*>(Bg + (size_t)row * ldb + ch * 8);
    }
  }
}

// -------- TDM-staged, double-buffered GEMM: used for the big compute-bound GEMMs --------
template <int EPI>
__global__ __launch_bounds__(256, 2)
void gemm_bf16_wmma_staged(const bf16_t* __restrict__ A, int lda,
                           const bf16_t* __restrict__ Bt, int ldb,
                           float* __restrict__ Cf, bf16_t* __restrict__ Cb, int ldc,
                           const float* __restrict__ bias,
                           const float* __restrict__ resid, int ldr,
                           int Kdim, int Mtot, int Ntot) {
  __shared__ bf16_t smem[2 * 128 * LDS_PITCH + 2 * 64 * LDS_PITCH];  // 30720 B
  bf16_t* const A0 = smem;
  bf16_t* const A1 = smem + 128 * LDS_PITCH;
  bf16_t* const B0 = smem + 2 * 128 * LDS_PITCH;
  bf16_t* const B1 = B0 + 64 * LDS_PITCH;

  const int lane = threadIdx.x & 31;
  const int wave = threadIdx.x >> 5;
  const int wm = wave & 3, wn = wave >> 2;
  const int rsel = lane & 15;
  const int g = lane >> 4;
  const int mblk = blockIdx.y * 128;
  const int nblk = blockIdx.x * 64;

  const bf16_t* Ag = A + (size_t)mblk * lda;
  const bf16_t* Bg = Bt + (size_t)nblk * ldb;
  const int nk = Kdim >> 5;

#if HAVE_TDM
  if (wave == 0) {   // one TDM descriptor pair per stage, issued by wave 0 only
    tdm_load_tile((unsigned)(uintptr_t)A0, Ag, 128, Kdim, Mtot - mblk, lda);
    tdm_load_tile((unsigned)(uintptr_t)B0, Bg, 64, Kdim, Ntot - nblk, ldb);
  }
#endif

  v8f acc00 = {}, acc01 = {}, acc10 = {}, acc11 = {};
  for (int ks = 0; ks < nk; ++ks) {
    const int p = ks & 1;
    bf16_t* const As = p ? A1 : A0;
    bf16_t* const Bs = p ? B1 : B0;
#if HAVE_TDM
    if (wave == 0) {
      if (ks + 1 < nk) {       // prefetch next stage into the other buffer
        tdm_load_tile((unsigned)(uintptr_t)(p ? A0 : A1), Ag + (size_t)(ks + 1) * 32, 128, Kdim, Mtot - mblk, lda);
        tdm_load_tile((unsigned)(uintptr_t)(p ? B0 : B1), Bg + (size_t)(ks + 1) * 32, 64, Kdim, Ntot - nblk, ldb);
        __builtin_amdgcn_s_wait_tensorcnt(2);  // TDM completes in order: current stage done
      } else {
        __builtin_amdgcn_s_wait_tensorcnt(0);
      }
    }
    __syncthreads();           // publish stage ks tiles to all waves
#else
    __syncthreads();
    stage_tiles_sync(As, Bs, Ag + (size_t)ks * 32, Bg + (size_t)ks * 32, lda, ldb, threadIdx.x);
    __syncthreads();
#endif
    const bf16_t* ap = As + (wm * 32 + rsel) * LDS_PITCH + g * 8;
    const bf16_t* bp = Bs + (wn * 32 + rsel) * LDS_PITCH + g * 8;
    v16bf fa0 = load_frag_a(ap);
    v16bf fa1 = load_frag_a(ap + 16 * LDS_PITCH);
    v16bf fb0 = load_frag_b(bp);
    v16bf fb1 = load_frag_b(bp + 16 * LDS_PITCH);
    acc00 = __builtin_amdgcn_wmma_f32_16x16x32_bf16(false, fa0, false, fb0, (short)0, acc00, false, false);
    acc01 = __builtin_amdgcn_wmma_f32_16x16x32_bf16(false, fa0, false, fb1, (short)0, acc01, false, false);
    acc10 = __builtin_amdgcn_wmma_f32_16x16x32_bf16(false, fa1, false, fb0, (short)0, acc10, false, false);
    acc11 = __builtin_amdgcn_wmma_f32_16x16x32_bf16(false, fa1, false, fb1, (short)0, acc11, false, false);
#if HAVE_TDM
    __syncthreads();           // everyone done reading before this buffer is refilled
#endif
  }
  v8f accs[2][2] = {{acc00, acc01}, {acc10, acc11}};
  gemm_epilogue<EPI>(accs, mblk + wm * 32, nblk + wn * 32, g, rsel, Cf, Cb, ldc, bias, resid, ldr);
}

// ---------------- LayerNorm (1024 wide) -> bf16 ----------------
__global__ __launch_bounds__(256)
void layernorm_bf16(const float* __restrict__ in, const float* __restrict__ gamma,
                    const float* __restrict__ beta, bf16_t* __restrict__ outp) {
  __shared__ float red[256];
  const int tid = threadIdx.x;
  const size_t base = (size_t)blockIdx.x * DMODEL;
  float4 xv = reinterpret_cast<const float4*>(in + base)[tid];
  red[tid] = xv.x + xv.y + xv.z + xv.w;
  __syncthreads();
  for (int off = 128; off > 0; off >>= 1) { if (tid < off) red[tid] += red[tid + off]; __syncthreads(); }
  const float mean = red[0] * (1.f / DMODEL);
  __syncthreads();
  const float d0 = xv.x - mean, d1 = xv.y - mean, d2 = xv.z - mean, d3 = xv.w - mean;
  red[tid] = d0 * d0 + d1 * d1 + d2 * d2 + d3 * d3;
  __syncthreads();
  for (int off = 128; off > 0; off >>= 1) { if (tid < off) red[tid] += red[tid + off]; __syncthreads(); }
  const float rstd = rsqrtf(red[0] * (1.f / DMODEL) + 1e-5f);
  const int i0 = tid * 4;
  outp[base + i0 + 0] = (bf16_t)(d0 * rstd * gamma[i0 + 0] + beta[i0 + 0]);
  outp[base + i0 + 1] = (bf16_t)(d1 * rstd * gamma[i0 + 1] + beta[i0 + 1]);
  outp[base + i0 + 2] = (bf16_t)(d2 * rstd * gamma[i0 + 2] + beta[i0 + 2]);
  outp[base + i0 + 3] = (bf16_t)(d3 * rstd * gamma[i0 + 3] + beta[i0 + 3]);
}

// ---------------- weight prep ----------------
__global__ __launch_bounds__(256)
void cast_scale_bf16(const float* __restrict__ in, bf16_t* __restrict__ out, int n, float scale) {
  for (int i = blockIdx.x * 256 + threadIdx.x; i < n; i += gridDim.x * 256)
    out[i] = (bf16_t)(in[i] * scale);
}

// out[c*ldout + r] = in[r*ldin + c] * scale ; tile = 32x32, block = 256 (32x8)
__global__ __launch_bounds__(256)
void transpose_cast(const float* __restrict__ in, int ldin,
                    bf16_t* __restrict__ out, int ldout, float scale) {
  __shared__ float tile[32][33];
  const int tx = threadIdx.x & 31, ty = threadIdx.x >> 5;
  const int c0 = blockIdx.x * 32, r0 = blockIdx.y * 32;
#pragma unroll
  for (int i = 0; i < 32; i += 8)
    tile[ty + i][tx] = in[(size_t)(r0 + ty + i) * ldin + c0 + tx] * scale;
  __syncthreads();
#pragma unroll
  for (int i = 0; i < 32; i += 8)
    out[(size_t)(c0 + ty + i) * ldout + r0 + tx] = (bf16_t)tile[tx][ty + i];
}

// BD = blockdiag(R^T, R^T), R = I + eps*M ; 128x128 bf16
__global__ __launch_bounds__(256)
void build_bd(const float* __restrict__ Mres, const float* __restrict__ epsP, bf16_t* __restrict__ BD) {
  const int idx = blockIdx.x * 256 + threadIdx.x;  // 0..16383
  const int i = idx >> 7, j = idx & 127;
  float v = 0.f;
  if ((i < 64) == (j < 64)) {
    const int ii = i & 63, jj = j & 63;
    v = (ii == jj ? 1.f : 0.f) + (*epsP) * Mres[jj * KD + ii];  // R^T[ii][jj] = R[jj][ii]
  }
  BD[idx] = (bf16_t)v;
}

// ---------------- chunked diagonal complex scan ----------------
__device__ __forceinline__ void get_lambda(const float* __restrict__ alpha,
                                           const float* __restrict__ omega,
                                           int k, float& lr, float& li) {
  const float mag = 1.f / (1.f + expf(-alpha[k]));
  lr = mag * cosf(omega[k]);
  li = mag * sinf(omega[k]);
}

__global__ __launch_bounds__(256)
void scan_phase1(const float* __restrict__ beta, const float* __restrict__ alpha,
                 const float* __restrict__ omega, float2* __restrict__ carr) {
  const int idx = blockIdx.x * 256 + threadIdx.x;  // 8192 threads
  const int k = idx & 63, bc = idx >> 6, c = bc & 31, b = bc >> 5;
  float lr, li; get_lambda(alpha, omega, k, lr, li);
  const float* bp = beta + ((size_t)(b * TLEN + c * CHLEN)) * 128 + k;
  float cr = 0.f, ci = 0.f;
  for (int j = 0; j < CHLEN; ++j) {
    const float br = bp[0], bi = bp[64];
    const float nr = lr * cr - li * ci + br;
    const float ni = lr * ci + li * cr + bi;
    cr = nr; ci = ni; bp += 128;
  }
  carr[((b * 64 + k) * NCH) + c] = make_float2(cr, ci);
}

__global__ __launch_bounds__(256)
void scan_phase2(const float* __restrict__ state, const float* __restrict__ alpha,
                 const float* __restrict__ omega, const float2* __restrict__ carr,
                 float2* __restrict__ inc, float* __restrict__ final_out) {
  const int bk = threadIdx.x;  // 0..255
  const int k = bk & 63, b = bk >> 6;
  float lr, li; get_lambda(alpha, omega, k, lr, li);
  float pr = lr, pi = li;                       // lambda^128 via 7 squarings
#pragma unroll
  for (int s = 0; s < 7; ++s) { const float nr = pr * pr - pi * pi, ni = 2.f * pr * pi; pr = nr; pi = ni; }
  float sr = state[(b * KD + k) * 2 + 0], si = state[(b * KD + k) * 2 + 1];
  for (int c = 0; c < NCH; ++c) {
    inc[bk * NCH + c] = make_float2(sr, si);
    const float2 cf = carr[bk * NCH + c];
    const float nr = pr * sr - pi * si + cf.x;
    const float ni = pr * si + pi * sr + cf.y;
    sr = nr; si = ni;
  }
  final_out[(b * KD + k) * 2 + 0] = sr;
  final_out[(b * KD + k) * 2 + 1] = si;
}

__global__ __launch_bounds__(256)
void scan_phase3(const float* __restrict__ beta, const float* __restrict__ alpha,
                 const float* __restrict__ omega, const float2* __restrict__ inc,
                 bf16_t* __restrict__ S) {
  const int idx = blockIdx.x * 256 + threadIdx.x;
  const int k = idx & 63, bc = idx >> 6, c = bc & 31, b = bc >> 5;
  float lr, li; get_lambda(alpha, omega, k, lr, li);
  const float2 s0 = inc[((b * 64 + k) * NCH) + c];
  float cr = s0.x, ci = s0.y;
  const size_t off = ((size_t)(b * TLEN + c * CHLEN)) * 128 + k;
  const float* bp = beta + off;
  bf16_t* sp = S + off;
  for (int j = 0; j < CHLEN; ++j) {
    const float br = bp[0], bi = bp[64];
    const float nr = lr * cr - li * ci + br;
    const float ni = lr * ci + li * cr + bi;
    cr = nr; ci = ni;
    sp[0] = (bf16_t)cr; sp[64] = (bf16_t)ci;
    bp += 128; sp += 128;
  }
}

// ---------------- launcher ----------------
extern "C" void kernel_launch(void* const* d_in, const int* in_sizes, int n_in,
                              void* d_out, int out_size, void* d_ws, size_t ws_size,
                              hipStream_t stream) {
  (void)in_sizes; (void)n_in; (void)out_size; (void)ws_size;
  const float* x     = (const float*)d_in[0];
  const float* state = (const float*)d_in[1];
  const float* Er    = (const float*)d_in[2];
  const float* Ei    = (const float*)d_in[3];
  const float* alpha = (const float*)d_in[4];
  const float* omega = (const float*)d_in[5];
  const float* epsP  = (const float*)d_in[6];
  const float* Mres  = (const float*)d_in[7];
  const float* out_w = (const float*)d_in[8];
  const float* out_b = (const float*)d_in[9];
  const float* w1    = (const float*)d_in[10];
  const float* b1    = (const float*)d_in[11];
  const float* w2    = (const float*)d_in[12];
  const float* b2    = (const float*)d_in[13];
  const float* g1    = (const float*)d_in[14];
  const float* be1   = (const float*)d_in[15];
  const float* g2    = (const float*)d_in[16];
  const float* be2   = (const float*)d_in[17];

  float* out = (float*)d_out;
  float* final_state = out + (size_t)BT_TOK * DMODEL;

  char* wp = (char*)d_ws;
  auto carve = [&](size_t bytes) -> char* {
    char* p = wp; wp += (bytes + 255) & ~(size_t)255; return p;
  };
  bf16_t* xn     = (bf16_t*)carve((size_t)BT_TOK * DMODEL * 2);  // LN1 out; reused for LN2
  bf16_t* Astk   = (bf16_t*)carve((size_t)128 * DMODEL * 2);     // [Er ; -Ei] bf16
  bf16_t* EhT    = (bf16_t*)carve((size_t)DMODEL * 128 * 2);     // Astk^T
  bf16_t* BD     = (bf16_t*)carve((size_t)128 * 128 * 2);        // blockdiag(R^T,R^T)
  float*  A2f    = (float*) carve((size_t)128 * DMODEL * 4);
  bf16_t* A2b    = (bf16_t*)carve((size_t)128 * DMODEL * 2);
  float*  Whf    = (float*) carve((size_t)128 * DMODEL * 4);     // folded W (128 x D)
  bf16_t* WhT    = (bf16_t*)carve((size_t)DMODEL * 128 * 2);
  bf16_t* owT    = (bf16_t*)carve((size_t)DMODEL * DMODEL * 2);
  bf16_t* w1T    = (bf16_t*)carve((size_t)MLPD * DMODEL * 2);
  bf16_t* w2T    = (bf16_t*)carve((size_t)DMODEL * MLPD * 2);
  float*  beta   = (float*) carve((size_t)BT_TOK * 128 * 4);     // [beta_r | beta_i]
  bf16_t* Sbf    = (bf16_t*)carve((size_t)BT_TOK * 128 * 2);     // scan states bf16
  float2* carr   = (float2*)carve((size_t)256 * NCH * sizeof(float2));
  float2* inc    = (float2*)carve((size_t)256 * NCH * sizeof(float2));
  float*  ybuf   = (float*) carve((size_t)BT_TOK * DMODEL * 4);
  bf16_t* hidden = (bf16_t*)carve((size_t)BT_TOK * MLPD * 2);

  const dim3 blk(256);

  // --- per-launch weight prep (bf16 casts + N-major transposes) ---
  cast_scale_bf16<<<dim3(256), blk, 0, stream>>>(Er, Astk, KD * DMODEL, 1.f);
  cast_scale_bf16<<<dim3(256), blk, 0, stream>>>(Ei, Astk + (size_t)KD * DMODEL, KD * DMODEL, -1.f);
  transpose_cast<<<dim3(DMODEL / 32, KD / 32), blk, 0, stream>>>(Er, DMODEL, EhT, 128, 1.f);
  transpose_cast<<<dim3(DMODEL / 32, KD / 32), blk, 0, stream>>>(Ei, DMODEL, EhT + 64, 128, -1.f);
  build_bd<<<dim3(64), blk, 0, stream>>>(Mres, epsP, BD);
  transpose_cast<<<dim3(DMODEL / 32, DMODEL / 32), blk, 0, stream>>>(out_w, DMODEL, owT, DMODEL, 1.f);
  transpose_cast<<<dim3(MLPD / 32, DMODEL / 32), blk, 0, stream>>>(w1, MLPD, w1T, DMODEL, 1.f);
  transpose_cast<<<dim3(DMODEL / 32, MLPD / 32), blk, 0, stream>>>(w2, DMODEL, w2T, MLPD, 1.f);

  // Fold blockdiag(R^T)·[Er;-Ei]·out_w into one 128 x D weight (tiny GEMMs, direct path)
  gemm_bf16_wmma<EPI_F32><<<dim3(DMODEL / 64, 1), blk, 0, stream>>>(
      BD, 128, EhT, 128, A2f, nullptr, DMODEL, nullptr, nullptr, 0, 128);
  cast_scale_bf16<<<dim3(256), blk, 0, stream>>>(A2f, A2b, 128 * DMODEL, 1.f);
  gemm_bf16_wmma<EPI_F32><<<dim3(DMODEL / 64, 1), blk, 0, stream>>>(
      A2b, DMODEL, owT, DMODEL, Whf, nullptr, DMODEL, nullptr, nullptr, 0, DMODEL);
  transpose_cast<<<dim3(DMODEL / 32, 128 / 32), blk, 0, stream>>>(Whf, DMODEL, WhT, 128, 1.f);

  // LN1
  layernorm_bf16<<<dim3(BT_TOK), blk, 0, stream>>>(x, g1, be1, xn);

  // beta_r / beta_i projections into interleaved (BT x 128) buffer (TDM-staged)
  gemm_bf16_wmma_staged<EPI_F32><<<dim3(1, BT_TOK / 128), blk, 0, stream>>>(
      xn, DMODEL, Astk, DMODEL, beta, nullptr, 128, nullptr, nullptr, 0, DMODEL, BT_TOK, 64);
  gemm_bf16_wmma_staged<EPI_F32><<<dim3(1, BT_TOK / 128), blk, 0, stream>>>(
      xn, DMODEL, Astk + (size_t)KD * DMODEL, DMODEL, beta + 64, nullptr, 128, nullptr, nullptr, 0, DMODEL, BT_TOK, 64);

  // chunked diagonal scan (8192-way parallel) + final state
  scan_phase1<<<dim3(32), blk, 0, stream>>>(beta, alpha, omega, carr);
  scan_phase2<<<dim3(1), blk, 0, stream>>>(state, alpha, omega, carr, inc, final_state);
  scan_phase3<<<dim3(32), blk, 0, stream>>>(beta, alpha, omega, inc, Sbf);

  // y = x + S @ W + out_b (K=128: direct path)
  gemm_bf16_wmma<EPI_BIAS_RESID_F32><<<dim3(DMODEL / 64, BT_TOK / 128), blk, 0, stream>>>(
      Sbf, 128, WhT, 128, ybuf, nullptr, DMODEL, out_b, x, DMODEL, 128);

  // LN2
  layernorm_bf16<<<dim3(BT_TOK), blk, 0, stream>>>(ybuf, g2, be2, xn);

  // MLP: hidden = gelu(yn @ w1 + b1) ; out = y + hidden @ w2 + b2 (TDM-staged)
  gemm_bf16_wmma_staged<EPI_BIAS_GELU_BF16><<<dim3(MLPD / 64, BT_TOK / 128), blk, 0, stream>>>(
      xn, DMODEL, w1T, DMODEL, nullptr, hidden, MLPD, b1, nullptr, 0, DMODEL, BT_TOK, MLPD);
  gemm_bf16_wmma_staged<EPI_BIAS_RESID_F32><<<dim3(DMODEL / 64, BT_TOK / 128), blk, 0, stream>>>(
      hidden, MLPD, w2T, MLPD, out, nullptr, DMODEL, b2, ybuf, DMODEL, MLPD, BT_TOK, DMODEL);
}
